// AnisotropicDiffusionKernel_33775622815868
// MI455X (gfx1250) — matrix operand order repaired and checked by reading, not compiled, not run on hardware
//
#include <hip/hip_runtime.h>

// ---------------------------------------------------------------------------
// Fused anisotropic diffusion step for MI455X (gfx1250, wave32, WMMA).
//   B=8, C=64, H=256, W=256, HID=64.
//   Per workgroup (256 thr = 8 wave32): one (b, h, 128-col segment).
//   Stencils (Sobel/Laplacian) in f32 from an LDS-staged x tile; the two 1x1
//   convs run as f16 WMMA GEMMs: [16 px x 128] x [128 x 64], then
//   [16 px x 64] x [64 x 64]. Output goes through LDS for coalesced stores.
// ---------------------------------------------------------------------------

typedef __attribute__((ext_vector_type(16))) _Float16 v16h;
typedef __attribute__((ext_vector_type(8)))  float    v8f;

#define C_CH   64
#define HID_CH 64
#define HH     256
#define WW     256
#define WTILE  128           // pixels per workgroup along W
#define NWAVES 8             // 8 waves x 16 px = 128
#define XS_PITCH 132         // 130 used (128 + 2 halo), padded
#define HST_PITCH 72         // 64 + 8 pad (halves) -> conflict-free transpose
#define OUT_PITCH 130        // 128 + 2 pad (floats)

// LDS byte layout (dynamic, ~173.5 KB total)
#define XS_BYTES   (C_CH * 3 * XS_PITCH * 4)                 // 101376
#define W1H_BYTES  (HID_CH * 2 * C_CH * 2)                   // 16384
#define W2H_BYTES  (C_CH * HID_CH * 2)                       // 8192
#define HST_BYTES  (NWAVES * 16 * HST_PITCH * 2)             // 18432
#define OST_BYTES  (C_CH * OUT_PITCH * 4)                    // 33280
#define SMEM_BYTES (XS_BYTES + W1H_BYTES + W2H_BYTES + HST_BYTES + OST_BYTES)

__device__ __forceinline__ float sigmoid_f(float v) {
  return 1.0f / (1.0f + __expf(-v));
}

__global__ void __launch_bounds__(256, 1)
aniso_diffusion_fused(const float* __restrict__ x,
                      const float* __restrict__ mask,
                      const float* __restrict__ w1,
                      const float* __restrict__ b1,
                      const float* __restrict__ w2,
                      const float* __restrict__ b2,
                      const float* __restrict__ rate_p,
                      float* __restrict__ out)
{
  extern __shared__ __align__(16) char smem[];
  float*    xs  = (float*)smem;                        // [C][3][XS_PITCH] f32
  _Float16* w1h = (_Float16*)(smem + XS_BYTES);        // [HID][2C] f16
  _Float16* w2h = (_Float16*)(smem + XS_BYTES + W1H_BYTES);
  _Float16* hst = (_Float16*)(smem + XS_BYTES + W1H_BYTES + W2H_BYTES);
  float*    ost = (float*)(smem + XS_BYTES + W1H_BYTES + W2H_BYTES + HST_BYTES);

  const int tid = threadIdx.x;
  const int bid = blockIdx.x;            // 8*256*2 = 4096 blocks
  const int b   = bid >> 9;              // / (256 rows * 2 segs)
  const int rem = bid & 511;
  const int h   = rem >> 1;
  const int w0  = (rem & 1) * WTILE;

  // ---- Stage x tile (3 rows, 130 halo cols, 64 ch) as f32; convert weights.
  const int NX = C_CH * 3 * 130;
  for (int i = tid; i < NX; i += 256) {
    int c  = i / (3 * 130);
    int r2 = i % (3 * 130);
    int rr = r2 / 130;                   // 0..2 -> h-1..h+1
    int cc = r2 % 130;                   // 0..129 -> w0-1..w0+128
    int gh = h + rr - 1;
    int gw = w0 + cc - 1;
    float v = 0.0f;
    if (gh >= 0 && gh < HH && gw >= 0 && gw < WW)
      v = x[(((size_t)b * C_CH + c) * HH + gh) * WW + gw];
    xs[(c * 3 + rr) * XS_PITCH + cc] = v;
  }
  for (int i = tid; i < HID_CH * 2 * C_CH; i += 256) w1h[i] = (_Float16)w1[i];
  for (int i = tid; i < C_CH * HID_CH;    i += 256) w2h[i] = (_Float16)w2[i];
  __syncthreads();

  const int lane = tid & 31;
  const int wv   = tid >> 5;             // wave id 0..7 -> pixel tile
  const int pm   = lane & 15;            // A-role: pixel row M; B-role: col N
  const int grp  = lane >> 4;            // lane half selects K sub-block
  const int xcol = wv * 16 + pm + 1;     // column in xs (halo-shifted)

  // ---- Build A fragments: M=16 pixels, K=128 = [grad_mag(64); x(64)].
  // 16-bit A 16x32 layout: lane<16 -> K j<8:j, j>=8:j+8 ; lane>=16 -> +8.
  v16h a[4];
#pragma unroll
  for (int kt = 0; kt < 2; ++kt) {       // grad_mag channels
#pragma unroll
    for (int j = 0; j < 16; ++j) {
      int kk = (j < 8 ? j : j + 8) + 8 * grp;
      int c  = kt * 32 + kk;
      const float* p0 = &xs[(c * 3 + 0) * XS_PITCH + xcol];
      const float* p1 = p0 + XS_PITCH;
      const float* p2 = p1 + XS_PITCH;
      float gx = (p0[1] - p0[-1]) + 2.0f * (p1[1] - p1[-1]) + (p2[1] - p2[-1]);
      float gy = (p2[-1] - p0[-1]) + 2.0f * (p2[0] - p0[0]) + (p2[1] - p0[1]);
      a[kt][j] = (_Float16)__builtin_sqrtf(gx * gx + gy * gy + 1e-8f);
    }
  }
#pragma unroll
  for (int kt = 2; kt < 4; ++kt) {       // raw x channels
#pragma unroll
    for (int j = 0; j < 16; ++j) {
      int kk = (j < 8 ? j : j + 8) + 8 * grp;
      int c  = (kt - 2) * 32 + kk;
      a[kt][j] = (_Float16)xs[(c * 3 + 1) * XS_PITCH + xcol];
    }
  }

  // ---- Conv1: h = relu(W1 @ combined + b1). 4 N-tiles x 4 K-tiles WMMA.
  // B 32x16 layout: lane<16 -> K j<8:j, j>=8:j+8 ; lane>=16 -> +8; N = lane%16.
  v8f hacc[4];
#pragma unroll
  for (int nt = 0; nt < 4; ++nt) {
    float bv = b1[nt * 16 + pm];
    v8f cacc;
#pragma unroll
    for (int e = 0; e < 8; ++e) cacc[e] = bv;
#pragma unroll
    for (int kt = 0; kt < 4; ++kt) {
      const _Float16* wp = &w1h[(nt * 16 + pm) * (2 * C_CH) + kt * 32 + 8 * grp];
      v16h bf;
#pragma unroll
      for (int j = 0; j < 16; ++j) bf[j] = wp[j < 8 ? j : j + 8];
      cacc = __builtin_amdgcn_wmma_f32_16x16x32_f16(
          false, a[kt], false, bf, (short)0, cacc, false, false);
    }
    hacc[nt] = cacc;
  }

  // ---- ReLU + per-wave LDS transpose: C/D layout -> A layout for conv2.
  _Float16* hw = &hst[wv * 16 * HST_PITCH];
#pragma unroll
  for (int nt = 0; nt < 4; ++nt) {
#pragma unroll
    for (int r = 0; r < 8; ++r) {
      float v = hacc[nt][r];
      v = v > 0.0f ? v : 0.0f;
      hw[(r + 8 * grp) * HST_PITCH + nt * 16 + pm] = (_Float16)v;  // [pixel][hid]
    }
  }

  v16h a2[2];
#pragma unroll
  for (int kt = 0; kt < 2; ++kt) {
#pragma unroll
    for (int j = 0; j < 16; ++j) {
      int K = (j < 8 ? j : j + 8) + 8 * grp + 32 * kt;
      a2[kt][j] = hw[pm * HST_PITCH + K];
    }
  }

  // ---- Conv2: cond = sigmoid(W2 @ h + b2). 4 N-tiles x 2 K-tiles WMMA.
  v8f cacc2[4];
#pragma unroll
  for (int nt = 0; nt < 4; ++nt) {
    float bv = b2[nt * 16 + pm];
    v8f cacc;
#pragma unroll
    for (int e = 0; e < 8; ++e) cacc[e] = bv;
#pragma unroll
    for (int kt = 0; kt < 2; ++kt) {
      const _Float16* wp = &w2h[(nt * 16 + pm) * HID_CH + kt * 32 + 8 * grp];
      v16h bf;
#pragma unroll
      for (int j = 0; j < 16; ++j) bf[j] = wp[j < 8 ? j : j + 8];
      cacc = __builtin_amdgcn_wmma_f32_16x16x32_f16(
          false, a2[kt], false, bf, (short)0, cacc, false, false);
    }
    cacc2[nt] = cacc;
  }

  // ---- Tail: sigmoid, Laplacian (f32 from LDS), diffusion update, mask blend.
  // D layout: lane L, vgpr r -> pixel m = r + 8*(L/16), channel n = nt*16+L%16.
  float rate = sigmoid_f(rate_p[0]);
  float mv[8];
#pragma unroll
  for (int r = 0; r < 8; ++r) {
    int m = r + 8 * grp;
    mv[r] = mask[((size_t)b * HH + h) * WW + w0 + wv * 16 + m];
  }
#pragma unroll
  for (int nt = 0; nt < 4; ++nt) {
#pragma unroll
    for (int r = 0; r < 8; ++r) {
      int m   = r + 8 * grp;
      int c   = nt * 16 + pm;
      int col = wv * 16 + m;                       // 0..127 within segment
      const float* p1 = &xs[(c * 3 + 1) * XS_PITCH + col + 1];
      float ctr = p1[0];
      float lap = p1[-XS_PITCH] + p1[XS_PITCH] + p1[-1] + p1[1] - 4.0f * ctr;
      float cd  = sigmoid_f(cacc2[nt][r]);
      float v   = ctr + rate * cd * lap;
      float mk  = mv[r];
      v = v * (1.0f - mk) + ctr * mk;
      ost[c * OUT_PITCH + col] = v;
    }
  }
  __syncthreads();

  // ---- Coalesced NCHW store: 128 contiguous floats per channel-row.
  for (int i = tid; i < C_CH * WTILE; i += 256) {
    int c = i >> 7;
    int w = i & 127;
    out[(((size_t)b * C_CH + c) * HH + h) * WW + w0 + w] = ost[c * OUT_PITCH + w];
  }
}

extern "C" void kernel_launch(void* const* d_in, const int* in_sizes, int n_in,
                              void* d_out, int out_size, void* d_ws, size_t ws_size,
                              hipStream_t stream) {
  (void)in_sizes; (void)n_in; (void)d_ws; (void)ws_size; (void)out_size;
  const float* x    = (const float*)d_in[0];
  const float* mask = (const float*)d_in[1];
  const float* w1   = (const float*)d_in[2];
  const float* b1   = (const float*)d_in[3];
  const float* w2   = (const float*)d_in[4];
  const float* b2   = (const float*)d_in[5];
  const float* rate = (const float*)d_in[6];
  float* out = (float*)d_out;

  // ~173.5 KB dynamic LDS (CDNA5 WGP has 320 KB); raise the per-kernel cap.
  static_assert(SMEM_BYTES <= 320 * 1024, "LDS budget");
  hipFuncSetAttribute((const void*)aniso_diffusion_fused,
                      hipFuncAttributeMaxDynamicSharedMemorySize, SMEM_BYTES);

  dim3 grid(8 * 256 * 2);   // B * H * (W / WTILE)
  dim3 block(256);
  aniso_diffusion_fused<<<grid, block, SMEM_BYTES, stream>>>(
      x, mask, w1, b1, w2, b2, rate, out);
}
